// PairStr2Pair_44547400794146
// MI455X (gfx1250) — matrix operand order repaired and checked by reading, not compiled, not run on hardware
//
#include <hip/hip_runtime.h>

#define LDIM 384
#define DPAIR 128
#define NHEAD 4
#define DHID 32
#define DRBF 36
static constexpr size_t NPIX = (size_t)LDIM * LDIM;

typedef __attribute__((ext_vector_type(16))) __bf16 v16bf;
typedef __attribute__((ext_vector_type(8)))  __bf16 v8bf;
typedef __attribute__((ext_vector_type(8)))  float  v8f;

__device__ __forceinline__ __bf16 f2bf(float f) {
  unsigned u = __builtin_bit_cast(unsigned, f);
  unsigned r = (u + 0x7FFFu + ((u >> 16) & 1u)) >> 16;
  unsigned short s = (unsigned short)r;
  return __builtin_bit_cast(__bf16, s);
}
__device__ __forceinline__ float bf2f(__bf16 h) {
  unsigned short s = __builtin_bit_cast(unsigned short, h);
  unsigned u = ((unsigned)s) << 16;
  return __builtin_bit_cast(float, u);
}
__device__ __forceinline__ float wred_sum(float v) {
  #pragma unroll
  for (int m = 16; m; m >>= 1) v += __shfl_xor(v, m, 32);
  return v;
}
__device__ __forceinline__ float wred_max(float v) {
  #pragma unroll
  for (int m = 16; m; m >>= 1) v = fmaxf(v, __shfl_xor(v, m, 32));
  return v;
}

// ---- WMMA fragment loaders (16-bit A 16x32 / B 32x16 layouts per CDNA5 ISA) ----
// A: lane<16 holds row M=lane, elements 0..7 -> K0..7, 8..15 -> K16..23;
//    lane>=16 holds row M=lane-16, elements -> K8..15 / K24..31.
__device__ __forceinline__ v16bf load_a_frag(const __bf16* __restrict__ row, int k0, int half) {
  v8bf lo = *(const v8bf*)(row + k0 + half * 8);
  v8bf hi = *(const v8bf*)(row + k0 + 16 + half * 8);
  return __builtin_shufflevector(lo, hi, 0,1,2,3,4,5,6,7,8,9,10,11,12,13,14,15);
}
// B: lane<16 holds col N=lane, elements 0..7 -> K0..7, 8..15 -> K16..23; lane>=16 col N=lane-16, K8..15 / K24..31.
// 'row' points at the N-th row of an (N x K) row-major (i.e. pre-transposed) matrix.
__device__ __forceinline__ v16bf load_b_frag(const __bf16* __restrict__ row, int k0, int half) {
  v8bf lo = *(const v8bf*)(row + k0 + half * 8);
  v8bf hi = *(const v8bf*)(row + k0 + 16 + half * 8);
  return __builtin_shufflevector(lo, hi, 0,1,2,3,4,5,6,7,8,9,10,11,12,13,14,15);
}

// ---- rbf embedding: out = relu(feat@emb_w+emb_b)@proj_w+proj_b  (per pixel) ----
__global__ void k_rbf_embed(const float* __restrict__ feat, const float* __restrict__ emb_w,
                            const float* __restrict__ emb_b, const float* __restrict__ proj_w,
                            const float* __restrict__ proj_b, float* __restrict__ outp) {
  __shared__ float fin[DRBF];
  __shared__ float hid[DPAIR];
  size_t p = blockIdx.x;
  int t = threadIdx.x;
  if (t < DRBF) fin[t] = feat[p * DRBF + t];
  __syncthreads();
  float acc = emb_b[t];
  #pragma unroll
  for (int k2 = 0; k2 < DRBF; ++k2) acc = fmaf(fin[k2], emb_w[k2 * DPAIR + t], acc);
  hid[t] = fmaxf(acc, 0.f);
  __syncthreads();
  float o = proj_b[t];
  for (int k2 = 0; k2 < DPAIR; ++k2) o = fmaf(hid[k2], proj_w[k2 * DPAIR + t], o);
  outp[p * DPAIR + t] = o;
}

// ---- LayerNorm over 128 channels -> bf16, optional transposed read ----
__global__ void k_ln128(const float* __restrict__ src, __bf16* __restrict__ dst,
                        const float* __restrict__ w, const float* __restrict__ b, int trans) {
  int wid = threadIdx.x >> 5, lane = threadIdx.x & 31;
  size_t pix = (size_t)blockIdx.x * 8 + wid;
  int n = (int)(pix / LDIM), i = (int)(pix % LDIM);
  size_t sp = trans ? ((size_t)i * LDIM + n) : pix;
  const float4 xv = *(const float4*)(src + sp * DPAIR + lane * 4);
  float s  = xv.x + xv.y + xv.z + xv.w;
  float ss = xv.x * xv.x + xv.y * xv.y + xv.z * xv.z + xv.w * xv.w;
  s = wred_sum(s); ss = wred_sum(ss);
  float mean = s * (1.f / DPAIR);
  float var  = ss * (1.f / DPAIR) - mean * mean;
  float rstd = rsqrtf(var + 1e-5f);
  float xs[4] = {xv.x, xv.y, xv.z, xv.w};
  #pragma unroll
  for (int c = 0; c < 4; ++c) {
    int ch = lane * 4 + c;
    dst[pix * DPAIR + ch] = f2bf((xs[c] - mean) * rstd * w[ch] + b[ch]);
  }
}

// ---- bias head: bvec[i,j,h] = LN(rbf[sel(i,j)]) @ wb[:,h] ----
__global__ void k_biasproj(const float* __restrict__ rbf, const float* __restrict__ w,
                           const float* __restrict__ b, const float* __restrict__ wb,
                           float* __restrict__ bvec, int trans) {
  int wid = threadIdx.x >> 5, lane = threadIdx.x & 31;
  size_t pix = (size_t)blockIdx.x * 8 + wid;
  int i = (int)(pix / LDIM), j = (int)(pix % LDIM);
  size_t sp = trans ? ((size_t)j * LDIM + i) : pix;
  const float4 xv = *(const float4*)(rbf + sp * DPAIR + lane * 4);
  float s  = xv.x + xv.y + xv.z + xv.w;
  float ss = xv.x * xv.x + xv.y * xv.y + xv.z * xv.z + xv.w * xv.w;
  s = wred_sum(s); ss = wred_sum(ss);
  float mean = s * (1.f / DPAIR), var = ss * (1.f / DPAIR) - mean * mean;
  float rstd = rsqrtf(var + 1e-5f);
  float xs[4] = {xv.x, xv.y, xv.z, xv.w};
  float acc[NHEAD] = {0.f, 0.f, 0.f, 0.f};
  #pragma unroll
  for (int c = 0; c < 4; ++c) {
    int ch = lane * 4 + c;
    float y = (xs[c] - mean) * rstd * w[ch] + b[ch];
    #pragma unroll
    for (int h = 0; h < NHEAD; ++h) acc[h] = fmaf(y, wb[ch * NHEAD + h], acc[h]);
  }
  #pragma unroll
  for (int h = 0; h < NHEAD; ++h) acc[h] = wred_sum(acc[h]);
  if (lane == 0) {
    #pragma unroll
    for (int h = 0; h < NHEAD; ++h) bvec[pix * NHEAD + h] = acc[h];
  }
}

// ---- transpose+convert weight (K x N fp32) -> (N x K bf16) ----
__global__ void k_wtrans(const float* __restrict__ W, __bf16* __restrict__ Wt, int K, int N) {
  int id = blockIdx.x * 256 + threadIdx.x;
  if (id >= K * N) return;
  int nn = id / K, kk = id - nn * K;
  Wt[id] = f2bf(W[kk * N + nn]);
}

// ---- generic WMMA GEMM: C = A(MxK bf16) * Wt^T (Wt is NxK bf16); one wave = 16x16 tile ----
// mode 0: out_bf[row*N+col] = c*scale        mode 1: out_bf = sigmoid(c+bias)
// mode 2: out_bf = relu(c+bias)              mode 3: out_f[orig] = base_f[orig] + c + bias
// mode 4: transposed store out_bf[col*NPIX+row] = c*scale (one 16B vector store per lane)
__global__ void k_gemm(const __bf16* __restrict__ A, const __bf16* __restrict__ Wt,
                       const float* __restrict__ bias, __bf16* __restrict__ out_bf,
                       float* __restrict__ out_f, const float* __restrict__ base_f,
                       int K, int N, int mode, float scale, int trans_out) {
  int lane = threadIdx.x, l = lane & 15, half = lane >> 4;
  const __bf16* arow = A + ((size_t)blockIdx.x * 16 + l) * K;
  const __bf16* brow = Wt + ((size_t)blockIdx.y * 16 + l) * K;
  v8f c = {0.f, 0.f, 0.f, 0.f, 0.f, 0.f, 0.f, 0.f};
  for (int k0 = 0; k0 < K; k0 += 32) {
    v16bf a = load_a_frag(arow, k0, half);
    v16bf b = load_b_frag(brow, k0, half);
    c = __builtin_amdgcn_wmma_f32_16x16x32_bf16(false, a, false, b, (short)0, c, false, false);
  }
  int col = blockIdx.y * 16 + l;
  if (mode == 4) {
    v8bf pk;
    #pragma unroll
    for (int r = 0; r < 8; ++r) pk[r] = f2bf(c[r] * scale);
    size_t rowbase = (size_t)blockIdx.x * 16 + half * 8;
    *(v8bf*)(out_bf + (size_t)col * NPIX + rowbase) = pk;
    return;
  }
  float bs = bias ? bias[col] : 0.f;
  #pragma unroll
  for (int r = 0; r < 8; ++r) {
    size_t row = (size_t)blockIdx.x * 16 + half * 8 + r;
    float v = c[r];
    if (mode == 0) {
      out_bf[row * N + col] = f2bf(v * scale);
    } else if (mode == 1) {
      out_bf[row * N + col] = f2bf(1.f / (1.f + __expf(-(v + bs))));
    } else if (mode == 2) {
      out_bf[row * N + col] = f2bf(fmaxf(v + bs, 0.f));
    } else {
      int n = (int)(row / LDIM), i = (int)(row % LDIM);
      size_t orig = trans_out ? ((size_t)i * LDIM + n) : row;
      out_f[orig * N + col] = base_f[orig * N + col] + v + bs;
    }
  }
}

// ---- tied-attention logits: logits[h,i,j] = sum_n sum_k q[n,i,h,k]*k[n,j,h,k] + bvec[i,j,h] ----
__global__ void k_logits(const __bf16* __restrict__ q, const __bf16* __restrict__ kk,
                         const float* __restrict__ bvec, float* __restrict__ logits) {
  int lane = threadIdx.x, l = lane & 15, half = lane >> 4;
  int ti = blockIdx.x, tj = blockIdx.y, h = blockIdx.z;
  const __bf16* qb = q  + ((size_t)ti * 16 + l) * DPAIR + h * DHID;
  const __bf16* kb = kk + ((size_t)tj * 16 + l) * DPAIR + h * DHID;
  v8f c = {0.f, 0.f, 0.f, 0.f, 0.f, 0.f, 0.f, 0.f};
  for (int n = 0; n < LDIM; ++n) {
    size_t off = (size_t)n * LDIM * DPAIR;
    v16bf a = load_a_frag(qb + off, 0, half);
    v16bf b = load_b_frag(kb + off, 0, half);
    c = __builtin_amdgcn_wmma_f32_16x16x32_bf16(false, a, false, b, (short)0, c, false, false);
  }
  #pragma unroll
  for (int r = 0; r < 8; ++r) {
    int i = ti * 16 + half * 8 + r;
    int j = tj * 16 + l;
    size_t pij = (size_t)i * LDIM + j;
    logits[(size_t)h * NPIX + pij] = c[r] + bvec[pij * NHEAD + h];
  }
}

// ---- softmax over j for each (h,i); write bf16 attn[h][i][j] ----
__global__ void k_softmax(const float* __restrict__ logits, __bf16* __restrict__ attn) {
  int lane = threadIdx.x;
  size_t hi = blockIdx.x;  // h*L + i
  const float* row = logits + hi * LDIM;
  float vals[LDIM / 32];
  float m = -3.4e38f;
  #pragma unroll
  for (int t = 0; t < LDIM / 32; ++t) { vals[t] = row[lane + 32 * t]; m = fmaxf(m, vals[t]); }
  m = wred_max(m);
  float s = 0.f;
  #pragma unroll
  for (int t = 0; t < LDIM / 32; ++t) { vals[t] = __expf(vals[t] - m); s += vals[t]; }
  s = wred_sum(s);
  float inv = 1.f / s;
  #pragma unroll
  for (int t = 0; t < LDIM / 32; ++t) attn[hi * LDIM + lane + 32 * t] = f2bf(vals[t] * inv);
}

// ---- out[n,i,h,d] = (sum_j attn[h,i,j]*vT[h*32+d, n*L+j]) * g[n,i,h,d] -> gated bf16 ----
// vT is the transposed v: vT[ch][pix], so B-fragment loads are contiguous 16B vectors.
__global__ void k_attnv(const __bf16* __restrict__ attn, const __bf16* __restrict__ vT,
                        const __bf16* __restrict__ g, __bf16* __restrict__ gated) {
  int lane = threadIdx.x, l = lane & 15, half = lane >> 4;
  int ti = blockIdx.x, td = blockIdx.y;
  int n = blockIdx.z >> 2, h = blockIdx.z & 3;
  const __bf16* arow = attn + ((size_t)h * LDIM + ti * 16 + l) * LDIM;
  const __bf16* vrow = vT + ((size_t)(h * DHID + td * 16 + l)) * NPIX + (size_t)n * LDIM;
  v8f c = {0.f, 0.f, 0.f, 0.f, 0.f, 0.f, 0.f, 0.f};
  for (int j0 = 0; j0 < LDIM; j0 += 32) {
    v16bf a = load_a_frag(arow, j0, half);
    v16bf b = load_b_frag(vrow, j0, half);
    c = __builtin_amdgcn_wmma_f32_16x16x32_bf16(false, a, false, b, (short)0, c, false, false);
  }
  #pragma unroll
  for (int r = 0; r < 8; ++r) {
    int i = ti * 16 + half * 8 + r;
    int d = td * 16 + l;
    size_t idx = ((size_t)n * LDIM + i) * DPAIR + h * DHID + d;
    gated[idx] = f2bf(c[r] * bf2f(g[idx]));
  }
}

extern "C" void kernel_launch(void* const* d_in, const int* in_sizes, int n_in,
                              void* d_out, int out_size, void* d_ws, size_t ws_size,
                              hipStream_t stream) {
  (void)in_sizes; (void)n_in; (void)out_size; (void)ws_size;
  const float* pair   = (const float*)d_in[0];
  const float* rbff   = (const float*)d_in[1];
  const float* emb_w  = (const float*)d_in[2];
  const float* emb_b  = (const float*)d_in[3];
  const float* proj_w = (const float*)d_in[4];
  const float* proj_b = (const float*)d_in[5];
  const float* ff_ln_w = (const float*)d_in[30];
  const float* ff_ln_b = (const float*)d_in[31];
  const float* ff_w1   = (const float*)d_in[32];
  const float* ff_b1   = (const float*)d_in[33];
  const float* ff_w2   = (const float*)d_in[34];
  const float* ff_b2   = (const float*)d_in[35];
  float* out = (float*)d_out;

  char* ws = (char*)d_ws;
  size_t o = 0;
  auto take = [&](size_t bytes) { size_t r = o; o += (bytes + 255) & ~(size_t)255; return r; };
  float*  rbf    = (float*)(ws + take(NPIX * DPAIR * 4));
  __bf16* xb     = (__bf16*)(ws + take(NPIX * DPAIR * 2));   // LN(x) bf16, reused as gated
  __bf16* qb     = (__bf16*)(ws + take(NPIX * DPAIR * 2));
  __bf16* kb     = (__bf16*)(ws + take(NPIX * DPAIR * 2));   // contiguous after qb
  __bf16* vTb    = (__bf16*)(ws + take(NPIX * DPAIR * 2));   // transposed v: [ch][pix]
  __bf16* gb     = (__bf16*)(ws + take(NPIX * DPAIR * 2));
  float*  bvec   = (float*)(ws + take(NPIX * NHEAD * 4));
  float*  logits = (float*)(ws + take(NPIX * NHEAD * 4));
  __bf16* attn   = (__bf16*)(ws + take(NPIX * NHEAD * 2));
  __bf16* wt     = (__bf16*)(ws + take((size_t)256 * 256 * 2)); // transposed-weight scratch
  __bf16* ffh    = qb; // FFN hidden (NPIX x 256 bf16) reuses contiguous q+k regions

  const float SCALE = 0.17677669529663687f;  // 32^-0.5
  const float KSC   = 1.0f / (float)LDIM;

  dim3 b32(32), b256(256);
  unsigned pixBlocks = (unsigned)(NPIX / 8);
  unsigned mt = (unsigned)(NPIX / 16);

  k_rbf_embed<<<dim3((unsigned)NPIX), dim3(128), 0, stream>>>(rbff, emb_w, emb_b, proj_w, proj_b, rbf);

  auto gemm = [&](const __bf16* A, const float* W, int K, int N, int mode, float scale,
                  const float* bias, __bf16* obf, float* of, const float* base, int trans,
                  unsigned Mtiles) {
    k_wtrans<<<dim3((unsigned)((K * N + 255) / 256)), b256, 0, stream>>>(W, wt, K, N);
    k_gemm<<<dim3(Mtiles, (unsigned)(N / 16)), b32, 0, stream>>>(A, wt, bias, obf, of, base,
                                                                 K, N, mode, scale, trans);
  };

  auto axial = [&](const float* base_src, int trans, int wi) {
    const float* lnp_w = (const float*)d_in[wi + 0];
    const float* lnp_b = (const float*)d_in[wi + 1];
    const float* lnb_w = (const float*)d_in[wi + 2];
    const float* lnb_b = (const float*)d_in[wi + 3];
    const float* wq = (const float*)d_in[wi + 4];
    const float* wk = (const float*)d_in[wi + 5];
    const float* wv = (const float*)d_in[wi + 6];
    const float* wbm = (const float*)d_in[wi + 7];
    const float* wg = (const float*)d_in[wi + 8];
    const float* bg = (const float*)d_in[wi + 9];
    const float* wo = (const float*)d_in[wi + 10];
    const float* bo = (const float*)d_in[wi + 11];
    k_ln128<<<dim3(pixBlocks), b256, 0, stream>>>(base_src, xb, lnp_w, lnp_b, trans);
    k_biasproj<<<dim3(pixBlocks), b256, 0, stream>>>(rbf, lnb_w, lnb_b, wbm, bvec, trans);
    gemm(xb, wq, DPAIR, DPAIR, 0, SCALE, nullptr, qb,  nullptr, nullptr, 0, mt);
    gemm(xb, wk, DPAIR, DPAIR, 0, KSC,   nullptr, kb,  nullptr, nullptr, 0, mt);
    gemm(xb, wv, DPAIR, DPAIR, 4, 1.0f,  nullptr, vTb, nullptr, nullptr, 0, mt);
    gemm(xb, wg, DPAIR, DPAIR, 1, 1.0f,  bg,      gb,  nullptr, nullptr, 0, mt);
    k_logits<<<dim3(LDIM / 16, LDIM / 16, NHEAD), b32, 0, stream>>>(qb, kb, bvec, logits);
    k_softmax<<<dim3(NHEAD * LDIM), b32, 0, stream>>>(logits, attn);
    k_attnv<<<dim3(LDIM / 16, DHID / 16, LDIM * NHEAD), b32, 0, stream>>>(attn, vTb, gb, xb);
    gemm(xb, wo, DPAIR, DPAIR, 3, 1.0f, bo, nullptr, out, base_src, trans, mt);
  };

  axial(pair, 1, 6);   // row attention (transposed frame), r_* weights
  axial(out, 0, 18);   // col attention, c_* weights

  // FFN: out += relu(LN(out)@w1+b1)@w2+b2
  k_ln128<<<dim3(pixBlocks), b256, 0, stream>>>(out, xb, ff_ln_w, ff_ln_b, 0);
  gemm(xb, ff_w1, DPAIR, 2 * DPAIR, 2, 1.0f, ff_b1, ffh, nullptr, nullptr, 0, mt);
  gemm(ffh, ff_w2, 2 * DPAIR, DPAIR, 3, 1.0f, ff_b2, nullptr, out, out, 0, mt);
}
